// TokenAttnPrune_62354335204038
// MI455X (gfx1250) — matrix operand order repaired and checked by reading, not compiled, not run on hardware
//
#include <hip/hip_runtime.h>
#include <hip/hip_bf16.h>
#include <math.h>

#define B_DIM  64
#define N_DIM  577
#define C_DIM  1024
#define K_SEL  144
#define P_ROW  640          // padded rows for X/Q/K storage: 40 tiles of 16
#define KTILES 40           // key-tile count (P_ROW/16), multiple of 4
#define QTILES 37           // query tiles containing any valid row (592/16)
#define ATTN_SCALE 0.03125f // 1024^-0.5
#define TILE_STRIDE (16 * C_DIM)   // elements between adjacent 16-row tiles

typedef __bf16 bf16;
typedef __attribute__((ext_vector_type(16))) __bf16 v16bf;
typedef __attribute__((ext_vector_type(8)))  __bf16 v8bf;
typedef __attribute__((ext_vector_type(8)))  float  v8f;

__device__ __forceinline__ v8bf cvt8(const float* __restrict__ p)
{
    float4 f0 = *(const float4*)(p + 0);
    float4 f1 = *(const float4*)(p + 4);
    v8bf r;
    r[0] = (bf16)f0.x; r[1] = (bf16)f0.y; r[2] = (bf16)f0.z; r[3] = (bf16)f0.w;
    r[4] = (bf16)f1.x; r[5] = (bf16)f1.y; r[6] = (bf16)f1.z; r[7] = (bf16)f1.w;
    return r;
}

// ---------------------------------------------------------------------------
// Kernel 0a: Wq/Wk fp32 -> bf16 (one pass, 8 elems/thread)
// ---------------------------------------------------------------------------
__global__ __launch_bounds__(256)
void tap_cvt_w(const float* __restrict__ Wq, const float* __restrict__ Wk,
               bf16* __restrict__ Wqbf, bf16* __restrict__ Wkbf)
{
    const size_t e = (size_t)(blockIdx.x * 256 + threadIdx.x) * 8; // < C*C
    *(v8bf*)(Wqbf + e) = cvt8(Wq + e);
    *(v8bf*)(Wkbf + e) = cvt8(Wk + e);
}

// ---------------------------------------------------------------------------
// Kernel 0b: hidden fp32 -> bf16, padded to P_ROW rows per batch (pad = 0)
// ---------------------------------------------------------------------------
__global__ __launch_bounds__(256)
void tap_cvt_x(const float* __restrict__ x, bf16* __restrict__ Xbf)
{
    const size_t e   = (size_t)(blockIdx.x * 256 + threadIdx.x) * 8; // < B*P_ROW*C
    const size_t row = e / C_DIM;
    const size_t c   = e % C_DIM;
    const int    b   = (int)(row / P_ROW);
    const int    n   = (int)(row % P_ROW);
    v8bf r;
    if (n < N_DIM) {
        r = cvt8(x + ((size_t)(b * N_DIM + n)) * C_DIM + c);
    } else {
        #pragma unroll
        for (int i = 0; i < 8; ++i) r[i] = (bf16)0.0f;
    }
    *(v8bf*)(Xbf + e) = r;
}

// ---------------------------------------------------------------------------
// Kernel 1: Q and K projection, register-blocked, pure bf16 hot loop.
// One wave -> 16 rows x 64 cols of BOTH q and k (8 accumulators).
// All 8 B-fragments use ONE base pointer + immediate offsets (t * 32KB) and
// are loaded before the 8 independent WMMAs -> clause-able, no serialization.
// ---------------------------------------------------------------------------
__global__ __launch_bounds__(32)
void tap_proj_wmma(const bf16* __restrict__ Xbf,
                   const bf16* __restrict__ Wqbf, const float* __restrict__ bq,
                   const bf16* __restrict__ Wkbf, const float* __restrict__ bk,
                   bf16* __restrict__ Qout, bf16* __restrict__ Kout)
{
    const int lane = threadIdx.x;
    const int half = lane >> 4;     // 0: lanes 0-15, 1: lanes 16-31
    const int l    = lane & 15;

    const int mtile   = blockIdx.x;        // over (B*P_ROW)/16
    const int colbase = blockIdx.y * 64;   // over C/64

    // A-fragment rows come from the zero-padded bf16 copy: no branches.
    const bf16* aptr = Xbf + (size_t)(mtile * 16 + l) * C_DIM;
    // B-fragment bases: row (colbase + l) of W, lane-half K offset folded in.
    const bf16* wqb = Wqbf + (size_t)(colbase + l) * C_DIM + half * 16;
    const bf16* wkb = Wkbf + (size_t)(colbase + l) * C_DIM + half * 16;

    v8f accQ[4] = {};
    v8f accK[4] = {};

    for (int k0 = 0; k0 < C_DIM; k0 += 32) {
        __builtin_prefetch(aptr + k0 + 128, 0, 1);   // global_prefetch_b8
        // A in contiguous k order (same as B): lane<16 -> K {0..15}; lane>=16 -> K {16..31}
        v8bf alo = *(const v8bf*)(aptr + k0 + half * 16);
        v8bf ahi = *(const v8bf*)(aptr + k0 + half * 16 + 8);
        v16bf a;
        #pragma unroll
        for (int e = 0; e < 8; ++e) { a[e] = alo[e]; a[8 + e] = ahi[e]; }

        // Batch all 8 fragment loads (immediate offsets), then 8 WMMAs.
        v16bf fq[4], fk[4];
        #pragma unroll
        for (int t = 0; t < 4; ++t) {
            fq[t] = *(const v16bf*)(wqb + k0 + t * TILE_STRIDE);
            fk[t] = *(const v16bf*)(wkb + k0 + t * TILE_STRIDE);
        }
        #pragma unroll
        for (int t = 0; t < 4; ++t) {
            accQ[t] = __builtin_amdgcn_wmma_f32_16x16x32_bf16(
                          false, a, false, fq[t], (short)0, accQ[t], false, false);
            accK[t] = __builtin_amdgcn_wmma_f32_16x16x32_bf16(
                          false, a, false, fk[t], (short)0, accK[t], false, false);
        }
    }

    // C/D layout: lane -> N = l; VGPR r -> M = r + half*8
    #pragma unroll
    for (int t = 0; t < 4; ++t) {
        const int   col = colbase + t * 16 + l;
        const float bvq = bq[col];
        const float bvk = bk[col];
        #pragma unroll
        for (int r = 0; r < 8; ++r) {
            const size_t o = (size_t)(mtile * 16 + r + half * 8) * C_DIM + col;
            Qout[o] = (bf16)(accQ[t][r] + bvq);
            Kout[o] = (bf16)(accK[t][r] + bvk);
        }
    }
}

// ---------------------------------------------------------------------------
// Kernel 2: per (batch, 16-query tile): S = Q Kt (scaled), row softmax in LDS,
// emit this tile's contribution to the column sums (deterministic).
// Key tiles register-blocked 4-wide, one base pointer + immediate offsets.
// ---------------------------------------------------------------------------
__global__ __launch_bounds__(32)
void tap_attn_colsum(const bf16* __restrict__ Q, const bf16* __restrict__ K,
                     float* __restrict__ partial /* [B][QTILES][P_ROW] */)
{
    __shared__ float sS[16 * P_ROW];
    __shared__ float sInv[16];

    const int lane = threadIdx.x;
    const int half = lane >> 4;
    const int l    = lane & 15;
    const int b    = blockIdx.x;
    const int qt   = blockIdx.y;

    const bf16* arow  = Q + ((size_t)b * P_ROW + qt * 16 + l) * C_DIM;
    const bf16* kbase = K + (size_t)b * P_ROW * C_DIM;

    for (int ct0 = 0; ct0 < KTILES; ct0 += 4) {
        const bf16* bptr = kbase + (size_t)(ct0 * 16 + l) * C_DIM + half * 16;
        v8f acc[4] = {};
        for (int k0 = 0; k0 < C_DIM; k0 += 32) {
            v8bf lo = *(const v8bf*)(arow + k0 + half * 16);
            v8bf hi = *(const v8bf*)(arow + k0 + half * 16 + 8);
            v16bf a;
            #pragma unroll
            for (int e = 0; e < 8; ++e) { a[e] = lo[e]; a[8 + e] = hi[e]; }

            v16bf bm[4];
            #pragma unroll
            for (int t = 0; t < 4; ++t)
                bm[t] = *(const v16bf*)(bptr + k0 + t * TILE_STRIDE);
            #pragma unroll
            for (int t = 0; t < 4; ++t)
                acc[t] = __builtin_amdgcn_wmma_f32_16x16x32_bf16(
                             false, a, false, bm[t], (short)0, acc[t], false, false);
        }
        #pragma unroll
        for (int t = 0; t < 4; ++t)
            #pragma unroll
            for (int r = 0; r < 8; ++r)
                sS[(r + half * 8) * P_ROW + (ct0 + t) * 16 + l] =
                    acc[t][r] * ATTN_SCALE;
    }
    __syncthreads();

    // Row softmax: row = l, the two lane-halves split the columns.
    const int row = l;
    float mx = -3.0e38f;
    for (int c = half; c < N_DIM; c += 2) mx = fmaxf(mx, sS[row * P_ROW + c]);
    mx = fmaxf(mx, __shfl_xor(mx, 16, 32));
    float sum = 0.0f;
    for (int c = half; c < N_DIM; c += 2) {
        float e = __expf(sS[row * P_ROW + c] - mx);
        sS[row * P_ROW + c] = e;
        sum += e;
    }
    sum += __shfl_xor(sum, 16, 32);
    if (half == 0) {
        const bool qvalid = (qt * 16 + row) < N_DIM;   // mask padded query rows
        sInv[row] = qvalid ? (1.0f / sum) : 0.0f;
    }
    __syncthreads();

    // Column partial sums for this query tile (padded cols >= 577 ignored).
    float* prow = partial + ((size_t)b * QTILES + qt) * P_ROW;
    for (int c = lane; c < N_DIM; c += 32) {
        float v = 0.0f;
        #pragma unroll
        for (int r = 0; r < 16; ++r) v += sS[r * P_ROW + c] * sInv[r];
        prow[c] = v;
    }
}

// ---------------------------------------------------------------------------
// Kernel 3: reduce the 37 per-tile partials -> summed[B][P_ROW]
// ---------------------------------------------------------------------------
__global__ __launch_bounds__(256)
void tap_reduce(const float* __restrict__ partial, float* __restrict__ summed)
{
    const int i = blockIdx.x * 256 + threadIdx.x;      // over B*P_ROW
    if (i >= B_DIM * P_ROW) return;
    const int b = i / P_ROW, c = i % P_ROW;
    float s = 0.0f;
    if (c < N_DIM) {
        for (int qt = 0; qt < QTILES; ++qt)
            s += partial[((size_t)b * QTILES + qt) * P_ROW + c];
    }
    summed[i] = s;
}

// ---------------------------------------------------------------------------
// Kernel 4: exact top-144 per batch by rank (strict total order: value desc,
// index asc) -> exactly K_SEL winners, emitted in ascending index order.
// ---------------------------------------------------------------------------
__global__ __launch_bounds__(256)
void tap_topk(const float* __restrict__ summed, int* __restrict__ idx)
{
    __shared__ float v[N_DIM];
    __shared__ int   rnk[N_DIM];
    const int b = blockIdx.x;
    const int t = threadIdx.x;

    for (int i = t; i < N_DIM; i += 256) v[i] = summed[b * P_ROW + i];
    __syncthreads();

    for (int i = t; i < N_DIM; i += 256) {
        const float vi = v[i];
        int r = 0;
        for (int j = 0; j < N_DIM; ++j) {
            const float vj = v[j];
            r += (vj > vi) || (vj == vi && j < i);
        }
        rnk[i] = r;
    }
    __syncthreads();

    for (int i = t; i < N_DIM; i += 256) {
        if (rnk[i] < K_SEL) {
            int pos = 0;
            for (int j = 0; j < i; ++j) pos += (rnk[j] < K_SEL) ? 1 : 0;
            idx[b * K_SEL + pos] = i;
        }
    }
}

// ---------------------------------------------------------------------------
// Kernel 5: gather selected rows (exact fp32 copy), one block per output row.
// ---------------------------------------------------------------------------
__global__ __launch_bounds__(256)
void tap_gather(const float* __restrict__ x, const int* __restrict__ idx,
                float* __restrict__ out)
{
    const int row = blockIdx.x;            // [0, B*K_SEL)
    const int b   = row / K_SEL;
    const int i   = idx[row];
    const float4* src = (const float4*)(x + (size_t)(b * N_DIM + i) * C_DIM);
    float4*       dst = (float4*)(out + (size_t)row * C_DIM);
    dst[threadIdx.x] = src[threadIdx.x];   // 256 * float4 == 1024 floats
}

// ---------------------------------------------------------------------------
extern "C" void kernel_launch(void* const* d_in, const int* in_sizes, int n_in,
                              void* d_out, int out_size, void* d_ws, size_t ws_size,
                              hipStream_t stream)
{
    const float* hidden = (const float*)d_in[0];
    const float* Wq     = (const float*)d_in[1];
    const float* bq     = (const float*)d_in[2];
    const float* Wk     = (const float*)d_in[3];
    const float* bk     = (const float*)d_in[4];
    // d_in[5] = k_select (compile-time K_SEL)

    // Workspace layout (all bf16 regions 16B-aligned by construction)
    const size_t xElems = (size_t)B_DIM * P_ROW * C_DIM;
    const size_t wElems = (size_t)C_DIM * C_DIM;
    bf16*  Xbf     = (bf16*)d_ws;
    bf16*  Wqbf    = Xbf + xElems;
    bf16*  Wkbf    = Wqbf + wElems;
    bf16*  Qbf     = Wkbf + wElems;
    bf16*  Kbf     = Qbf + xElems;
    float* partial = (float*)(Kbf + xElems);                     // B*QTILES*P_ROW
    float* summed  = partial + (size_t)B_DIM * QTILES * P_ROW;   // B*P_ROW
    int*   idx     = (int*)(summed + (size_t)B_DIM * P_ROW);     // B*K_SEL

    tap_cvt_w<<<(int)(wElems / 8 / 256), 256, 0, stream>>>(Wq, Wk, Wqbf, Wkbf);
    tap_cvt_x<<<(int)(xElems / 8 / 256), 256, 0, stream>>>(hidden, Xbf);

    dim3 g1((B_DIM * P_ROW) / 16, C_DIM / 64);
    tap_proj_wmma<<<g1, 32, 0, stream>>>(Xbf, Wqbf, bq, Wkbf, bk, Qbf, Kbf);

    dim3 g2(B_DIM, QTILES);
    tap_attn_colsum<<<g2, 32, 0, stream>>>(Qbf, Kbf, partial);

    const int nsum = B_DIM * P_ROW;
    tap_reduce<<<(nsum + 255) / 256, 256, 0, stream>>>(partial, summed);

    tap_topk<<<B_DIM, 256, 0, stream>>>(summed, idx);

    tap_gather<<<B_DIM * K_SEL, 256, 0, stream>>>(hidden, idx, (float*)d_out);
}
